// JambaAttentionDecoderLayer_67242007986399
// MI455X (gfx1250) — matrix operand order repaired and checked by reading, not compiled, not run on hardware
//
#include <hip/hip_runtime.h>
#include <hip/hip_bf16.h>

// ---------------------------------------------------------------------------
// Jamba attention+MoE decoder layer for MI455X (gfx1250, wave32, WMMA).
// bf16 WMMA (f32 accumulate); routed top-2 MoE grouped GEMM; software-
// pipelined double-buffered GEMM cores with async global->LDS A-tile copies
// and minimal-wait barriers (drain DScnt/ASYNCcnt only, keep global loads
// in flight across the WMMA phase).
// ---------------------------------------------------------------------------

#define DEVFN __device__ __forceinline__

typedef __attribute__((ext_vector_type(16))) __bf16 v16bf;
typedef __attribute__((ext_vector_type(8)))  float  v8f;
typedef __attribute__((ext_vector_type(4)))  int    v4i;

constexpr int T  = 1024, H = 1024, NH = 16, NKV = 4, HD = 64;
constexpr int I  = 2816, E = 8, SW = 512;
constexpr int QKVD = (NH + 2 * NKV) * HD;          // 1536
constexpr float EPS = 1e-6f;
constexpr float SCALE = 0.125f;                    // 64^-0.5

union FragU { v16bf v; uint4 q[2]; };

// A-matrix 16x32 bf16 fragment from row-major LDS tile (pitch in elements).
// lanes 0-15 hold K=0..15, lanes 16-31 hold K=16..31 (matches lds_bfrag).
DEVFN v16bf lds_afrag(const __bf16* base, int pitch, int lane) {
  int m = lane & 15, half = lane >> 4;
  FragU f;
  f.q[0] = *(const uint4*)(base + m * pitch + half * 16);
  f.q[1] = *(const uint4*)(base + m * pitch + half * 16 + 8);
  return f.v;
}
// B-matrix 32x16 (KxN) bf16 fragment from LDS tile stored [N][K] row-major.
// lanes 0-15 hold K=0..15, lanes 16-31 hold K=16..31, column = lane&15.
DEVFN v16bf lds_bfrag(const __bf16* base, int pitch, int lane) {
  int n = lane & 15, half = lane >> 4;
  const uint4* p = (const uint4*)(base + n * pitch + half * 16);
  FragU f; f.q[0] = p[0]; f.q[1] = p[1];
  return f.v;
}
DEVFN v8f wmma_bf16(v16bf a, v16bf b, v8f c) {
  return __builtin_amdgcn_wmma_f32_16x16x32_bf16(false, a, false, b,
                                                 (short)0, c, false, false);
}
DEVFN __bf16 tobf(float x) { return (__bf16)x; }

// ---- async global->LDS copy of one 16B chunk (CDNA5 ASYNCcnt path) --------
DEVFN void copy16_async(const __bf16* gsrc, __bf16* ldst) {
#if defined(__has_builtin) && __has_builtin(__builtin_amdgcn_global_load_async_to_lds_b128)
  __builtin_amdgcn_global_load_async_to_lds_b128(
      (__attribute__((address_space(1))) v4i*)gsrc,
      (__attribute__((address_space(3))) v4i*)ldst, 0, 0);
#else
  *(uint4*)ldst = *(const uint4*)gsrc;
#endif
}
DEVFN void wait_async0() {
#if defined(__has_builtin) && __has_builtin(__builtin_amdgcn_s_wait_asynccnt)
  __builtin_amdgcn_s_wait_asynccnt(0);
#else
  asm volatile("s_wait_asynccnt 0" ::: "memory");
#endif
}
DEVFN void wait_ds0() {
#if defined(__has_builtin) && __has_builtin(__builtin_amdgcn_s_wait_dscnt)
  __builtin_amdgcn_s_wait_dscnt(0);
#else
  asm volatile("s_wait_dscnt 0" ::: "memory");
#endif
}
// Barrier that drains only DScnt + ASYNCcnt (LDS visibility) but leaves
// global loads in flight — unlike __syncthreads(), whose release fence
// also drains LOADcnt and would serialize the software pipeline.
DEVFN void block_sync_lds() {
  wait_ds0();
  wait_async0();
  __builtin_amdgcn_s_barrier();
}

// ---- tile staging helpers (256 threads) -----------------------------------
// A tile: 64 rows x 32 k of bf16, one 16B chunk per thread (async copy).
DEVFN void copy_a_tile(const __bf16* Arow, size_t pitch, int kk, int tid,
                       __bf16* As) {
  int m = tid >> 2, q = tid & 3;
  copy16_async(&Arow[(size_t)m * pitch + kk + q * 8], &As[m * 32 + q * 8]);
}
// B tile: 128 rows x 32 k of f32 -> registers (4 float4 per thread).
DEVFN void load_b_tile(const float* Brow, size_t pitch, int kk, int tid,
                       float4 (&r)[4]) {
#pragma unroll
  for (int i = 0; i < 4; i++) {
    int li = i * 256 + tid, rr = li >> 3, q = li & 7;
    const float* src = &Brow[(size_t)rr * pitch + kk + q * 4];
    r[i] = *(const float4*)src;
    if ((q & 3) == 0) __builtin_prefetch(src + 32, 0, 0);  // next-next K tile
  }
}
// registers -> bf16 LDS tile [128][32]
DEVFN void store_b_tile(__bf16* Bs, int tid, const float4 (&r)[4]) {
#pragma unroll
  for (int i = 0; i < 4; i++) {
    int li = i * 256 + tid, rr = li >> 3, q = li & 7;
    __bf16* d = &Bs[rr * 32 + q * 4];
    d[0] = tobf(r[i].x); d[1] = tobf(r[i].y);
    d[2] = tobf(r[i].z); d[3] = tobf(r[i].w);
  }
}

// ---------------------------------------------------------------------------
__global__ void init_kernel(float* out_moe, int* counts) {
  int idx = blockIdx.x * 256 + threadIdx.x;
  if (idx < T * H) out_moe[idx] = 0.f;
  if (idx < E) counts[idx] = 0;
}

// ---------------------------------------------------------------------------
__global__ __launch_bounds__(256) void rmsnorm1_kernel(
    const float* __restrict__ x, const float* __restrict__ w,
    __bf16* __restrict__ out) {
  int row = blockIdx.x, tid = threadIdx.x;
  float v[4]; float ss = 0.f;
#pragma unroll
  for (int i = 0; i < 4; i++) {
    v[i] = x[(size_t)row * H + tid + i * 256];
    ss += v[i] * v[i];
  }
  __shared__ float red[8];
#pragma unroll
  for (int o = 16; o; o >>= 1) ss += __shfl_xor(ss, o, 32);
  if ((tid & 31) == 0) red[tid >> 5] = ss;
  __syncthreads();
  float tot = 0.f;
#pragma unroll
  for (int i = 0; i < 8; i++) tot += red[i];
  float inv = rsqrtf(tot / (float)H + EPS);
#pragma unroll
  for (int i = 0; i < 4; i++) {
    int c = tid + i * 256;
    out[(size_t)row * H + c] = tobf(v[i] * inv * w[c]);
  }
}

// ---------------------------------------------------------------------------
// residual = hs + attn_out (stored f32), h2 = rmsnorm(residual) (bf16)
__global__ __launch_bounds__(256) void add_rms_kernel(
    const float* __restrict__ hs, const float* __restrict__ ao,
    const float* __restrict__ w, float* __restrict__ resid,
    __bf16* __restrict__ h2) {
  int row = blockIdx.x, tid = threadIdx.x;
  float v[4]; float ss = 0.f;
#pragma unroll
  for (int i = 0; i < 4; i++) {
    int c = tid + i * 256;
    v[i] = hs[(size_t)row * H + c] + ao[(size_t)row * H + c];
    resid[(size_t)row * H + c] = v[i];
    ss += v[i] * v[i];
  }
  __shared__ float red[8];
#pragma unroll
  for (int o = 16; o; o >>= 1) ss += __shfl_xor(ss, o, 32);
  if ((tid & 31) == 0) red[tid >> 5] = ss;
  __syncthreads();
  float tot = 0.f;
#pragma unroll
  for (int i = 0; i < 8; i++) tot += red[i];
  float inv = rsqrtf(tot / (float)H + EPS);
#pragma unroll
  for (int i = 0; i < 4; i++) {
    int c = tid + i * 256;
    h2[(size_t)row * H + c] = tobf(v[i] * inv * w[c]);
  }
}

// ---------------------------------------------------------------------------
// C[M,N] = A_bf16[M,K] * B_f32[N,K]^T. BM=64 BN=128 BK=32, 8 waves, each
// wave a 32x32 sub-tile. Double-buffered LDS, B staged one K-step ahead.
__global__ __launch_bounds__(256) void gemm_kernel(
    const __bf16* __restrict__ A, const float* __restrict__ B,
    float* __restrict__ C, int M, int N, int K) {
  __shared__ __bf16 As[2][64 * 32];
  __shared__ __bf16 Bs[2][128 * 32];
  int tid = threadIdx.x, lane = tid & 31, wid = tid >> 5;
  int m0b = blockIdx.x * 64, n0b = blockIdx.y * 128;
  int wm = (wid & 1) * 32, wn = (wid >> 1) * 32;
  const __bf16* Arow = A + (size_t)m0b * K;
  const float*  Brow = B + (size_t)n0b * K;
  v8f acc[2][2] = {};
  float4 breg[4];
  load_b_tile(Brow, K, 0, tid, breg);
  int p = 0;
  for (int kk = 0; kk < K; kk += 32, p ^= 1) {
    copy_a_tile(Arow, K, kk, tid, As[p]);
    store_b_tile(Bs[p], tid, breg);
    if (kk + 32 < K) load_b_tile(Brow, K, kk + 32, tid, breg);
    block_sync_lds();
    v16bf a0 = lds_afrag(&As[p][wm * 32], 32, lane);
    v16bf a1 = lds_afrag(&As[p][(wm + 16) * 32], 32, lane);
    v16bf b0 = lds_bfrag(&Bs[p][wn * 32], 32, lane);
    v16bf b1 = lds_bfrag(&Bs[p][(wn + 16) * 32], 32, lane);
    acc[0][0] = wmma_bf16(a0, b0, acc[0][0]);
    acc[0][1] = wmma_bf16(a0, b1, acc[0][1]);
    acc[1][0] = wmma_bf16(a1, b0, acc[1][0]);
    acc[1][1] = wmma_bf16(a1, b1, acc[1][1]);
  }
  int half = lane >> 4, nidx = lane & 15;
#pragma unroll
  for (int mi = 0; mi < 2; mi++)
#pragma unroll
    for (int ni = 0; ni < 2; ni++)
#pragma unroll
      for (int r = 0; r < 8; r++)
        C[(size_t)(m0b + wm + mi * 16 + r + half * 8) * N +
          n0b + wn + ni * 16 + nidx] = acc[mi][ni][r];
}

// ---------------------------------------------------------------------------
// GQA sliding-window causal attention, flash-style. One wave per
// (head, 16-query block); 4 independent waves per 128-thread block.
__global__ __launch_bounds__(128) void attn_kernel(
    const float* __restrict__ qkv, __bf16* __restrict__ out) {
  __shared__ __bf16 Qt[4][16][64];
  __shared__ __bf16 Kt[4][32][64];
  __shared__ __bf16 Vt[4][64][32];   // transposed: [hd][key]
  __shared__ float  Sf[4][16][32];
  __shared__ __bf16 Pt[4][16][32];
  __shared__ float  Rw[4][16];
  int tid = threadIdx.x, lane = tid & 31, wid = tid >> 5;
  int head = blockIdx.y, kvh = head / (NH / NKV);
  int q0 = (blockIdx.x * 4 + wid) * 16;
  int qcol = head * HD;
  int kcol = NH * HD + kvh * HD;
  int vcol = (NH + NKV) * HD + kvh * HD;

  // stage Q tile (16x64) as bf16
#pragma unroll
  for (int i = 0; i < 8; i++) {
    int li = lane + i * 32, r = li >> 4, c4 = li & 15;
    float4 f = *(const float4*)&qkv[(size_t)(q0 + r) * QKVD + qcol + c4 * 4];
    __bf16* d = &Qt[wid][r][c4 * 4];
    d[0] = tobf(f.x); d[1] = tobf(f.y); d[2] = tobf(f.z); d[3] = tobf(f.w);
  }
  __builtin_amdgcn_wave_barrier();
  v16bf aq0 = lds_afrag(&Qt[wid][0][0], 64, lane);
  v16bf aq1 = lds_afrag(&Qt[wid][0][32], 64, lane);

  v8f acc[4] = {};
  float m_i = -1e30f, l_i = 0.f;
  int jstart = q0 - SW + 1; if (jstart < 0) jstart = 0; jstart &= ~31;
  for (int jb = jstart; jb <= q0 + 15; jb += 32) {
    // stage K (row-major) and V (transposed) blocks, 32 keys
#pragma unroll
    for (int i = 0; i < 16; i++) {
      int li = lane + i * 32, r = li >> 4, c4 = li & 15;
      int j = jb + r; if (j > T - 1) j = T - 1;
      float4 fk = *(const float4*)&qkv[(size_t)j * QKVD + kcol + c4 * 4];
      float4 fv = *(const float4*)&qkv[(size_t)j * QKVD + vcol + c4 * 4];
      __bf16* dk = &Kt[wid][r][c4 * 4];
      dk[0] = tobf(fk.x); dk[1] = tobf(fk.y); dk[2] = tobf(fk.z); dk[3] = tobf(fk.w);
      Vt[wid][c4 * 4 + 0][r] = tobf(fv.x);
      Vt[wid][c4 * 4 + 1][r] = tobf(fv.y);
      Vt[wid][c4 * 4 + 2][r] = tobf(fv.z);
      Vt[wid][c4 * 4 + 3][r] = tobf(fv.w);
    }
    __builtin_amdgcn_wave_barrier();
    // S = Q * K^T  (16x32)
#pragma unroll
    for (int nt = 0; nt < 2; nt++) {
      v8f s = {};
      s = wmma_bf16(aq0, lds_bfrag(&Kt[wid][nt * 16][0], 64, lane), s);
      s = wmma_bf16(aq1, lds_bfrag(&Kt[wid][nt * 16][32], 64, lane), s);
#pragma unroll
      for (int r = 0; r < 8; r++)
        Sf[wid][r + (lane >> 4) * 8][nt * 16 + (lane & 15)] = s[r];
    }
    __builtin_amdgcn_wave_barrier();
    // online softmax: one lane per query row
    if (lane < 16) {
      int q = q0 + lane;
      float mx = m_i;
#pragma unroll
      for (int c = 0; c < 32; c++) {
        int j = jb + c;
        bool ok = (j <= q) && (q - j < SW);
        float sv = ok ? Sf[wid][lane][c] * SCALE : -1e30f;
        Sf[wid][lane][c] = sv;
        mx = fmaxf(mx, sv);
      }
      float sum = 0.f;
#pragma unroll
      for (int c = 0; c < 32; c++) {
        float pv = __expf(Sf[wid][lane][c] - mx);
        sum += pv;
        Pt[wid][lane][c] = tobf(pv);
      }
      float resc = __expf(m_i - mx);
      l_i = l_i * resc + sum;
      m_i = mx;
      Rw[wid][lane] = resc;
    }
    __builtin_amdgcn_wave_barrier();
    // rescale accumulator, then O += P * V
    v16bf ap = lds_afrag(&Pt[wid][0][0], 32, lane);
#pragma unroll
    for (int nt = 0; nt < 4; nt++) {
#pragma unroll
      for (int r = 0; r < 8; r++) acc[nt][r] *= Rw[wid][r + (lane >> 4) * 8];
      acc[nt] = wmma_bf16(ap, lds_bfrag(&Vt[wid][nt * 16][0], 32, lane), acc[nt]);
    }
    __builtin_amdgcn_wave_barrier();
  }
  if (lane < 16) Rw[wid][lane] = 1.f / l_i;
  __builtin_amdgcn_wave_barrier();
#pragma unroll
  for (int nt = 0; nt < 4; nt++)
#pragma unroll
    for (int r = 0; r < 8; r++) {
      int row = q0 + r + (lane >> 4) * 8;
      int col = head * HD + nt * 16 + (lane & 15);
      out[(size_t)row * H + col] = tobf(acc[nt][r] * Rw[wid][r + (lane >> 4) * 8]);
    }
}

// ---------------------------------------------------------------------------
// Router: softmax over E logits, top-2, build per-expert token lists.
__global__ __launch_bounds__(256) void router_kernel(
    const __bf16* __restrict__ h2, const float* __restrict__ rw,
    int* __restrict__ counts, int* __restrict__ list,
    float* __restrict__ slotw) {
  int tid = threadIdx.x, lane = tid & 31, wid = tid >> 5;
  int t = blockIdx.x * 8 + wid;
  float acc[E] = {};
  for (int c = lane; c < H; c += 32) {
    float x = (float)h2[(size_t)t * H + c];
#pragma unroll
    for (int e = 0; e < E; e++) acc[e] += x * rw[e * H + c];
  }
#pragma unroll
  for (int e = 0; e < E; e++)
#pragma unroll
    for (int o = 16; o; o >>= 1) acc[e] += __shfl_xor(acc[e], o, 32);
  if (lane == 0) {
    float mx = acc[0];
#pragma unroll
    for (int e = 1; e < E; e++) mx = fmaxf(mx, acc[e]);
    float p[E]; float s = 0.f;
#pragma unroll
    for (int e = 0; e < E; e++) { p[e] = __expf(acc[e] - mx); s += p[e]; }
    float invs = 1.f / s;
    int i1 = 0;
#pragma unroll
    for (int e = 1; e < E; e++) if (p[e] > p[i1]) i1 = e;
    int i2 = (i1 == 0) ? 1 : 0;
#pragma unroll
    for (int e = 0; e < E; e++) if (e != i1 && p[e] > p[i2]) i2 = e;
    int pos1 = atomicAdd(&counts[i1], 1);
    list[i1 * T + pos1] = t; slotw[i1 * T + pos1] = p[i1] * invs;
    int pos2 = atomicAdd(&counts[i2], 1);
    list[i2 * T + pos2] = t; slotw[i2 * T + pos2] = p[i2] * invs;
  }
}

// ---------------------------------------------------------------------------
// MoE GEMM1: act[e][slot][I] = silu(h2_g @ Wg^T) * (h2_g @ Wu^T); gathered A,
// double-buffered LDS, two B streams staged one K-step ahead.
__global__ __launch_bounds__(256) void moe_gemm1_kernel(
    const __bf16* __restrict__ h2, const float* __restrict__ wsw,
    const int* __restrict__ counts, const int* __restrict__ list,
    __bf16* __restrict__ act) {
  int e = blockIdx.z;
  int cnt = counts[e];
  int m0b = blockIdx.x * 64;
  if (m0b >= cnt) return;
  int n0b = blockIdx.y * 128;
  __shared__ __bf16 As[2][64 * 32];
  __shared__ __bf16 Bg[2][128 * 32];
  __shared__ __bf16 Bu[2][128 * 32];
  __shared__ int rowTok[64];
  int tid = threadIdx.x, lane = tid & 31, wid = tid >> 5;
  if (tid < 64) {
    int s = m0b + tid;
    rowTok[tid] = list[e * T + (s < cnt ? s : 0)];
  }
  __syncthreads();
  int wm = (wid & 1) * 32, wn = (wid >> 1) * 32;
  const float* WgB = wsw + (size_t)e * 2 * I * H + (size_t)n0b * H;
  const float* WuB = wsw + (size_t)e * 2 * I * H + (size_t)(I + n0b) * H;
  int am = tid >> 2, aq = tid & 3;              // A chunk for this thread
  const __bf16* Asrc = &h2[(size_t)rowTok[am] * H + aq * 8];
  v8f ag[2][2] = {}, au[2][2] = {};
  float4 gr[4], ur[4];
  load_b_tile(WgB, H, 0, tid, gr);
  load_b_tile(WuB, H, 0, tid, ur);
  int p = 0;
  for (int kk = 0; kk < H; kk += 32, p ^= 1) {
    copy16_async(Asrc + kk, &As[p][am * 32 + aq * 8]);
    store_b_tile(Bg[p], tid, gr);
    store_b_tile(Bu[p], tid, ur);
    if (kk + 32 < H) {
      load_b_tile(WgB, H, kk + 32, tid, gr);
      load_b_tile(WuB, H, kk + 32, tid, ur);
    }
    block_sync_lds();
    v16bf a0 = lds_afrag(&As[p][wm * 32], 32, lane);
    v16bf a1 = lds_afrag(&As[p][(wm + 16) * 32], 32, lane);
    v16bf g0 = lds_bfrag(&Bg[p][wn * 32], 32, lane);
    v16bf g1 = lds_bfrag(&Bg[p][(wn + 16) * 32], 32, lane);
    v16bf u0 = lds_bfrag(&Bu[p][wn * 32], 32, lane);
    v16bf u1 = lds_bfrag(&Bu[p][(wn + 16) * 32], 32, lane);
    ag[0][0] = wmma_bf16(a0, g0, ag[0][0]);
    ag[0][1] = wmma_bf16(a0, g1, ag[0][1]);
    ag[1][0] = wmma_bf16(a1, g0, ag[1][0]);
    ag[1][1] = wmma_bf16(a1, g1, ag[1][1]);
    au[0][0] = wmma_bf16(a0, u0, au[0][0]);
    au[0][1] = wmma_bf16(a0, u1, au[0][1]);
    au[1][0] = wmma_bf16(a1, u0, au[1][0]);
    au[1][1] = wmma_bf16(a1, u1, au[1][1]);
  }
  int half = lane >> 4, nidx = lane & 15;
#pragma unroll
  for (int mi = 0; mi < 2; mi++)
#pragma unroll
    for (int ni = 0; ni < 2; ni++)
#pragma unroll
      for (int r = 0; r < 8; r++) {
        int slot = m0b + wm + mi * 16 + r + half * 8;
        if (slot < cnt) {
          int col = n0b + wn + ni * 16 + nidx;
          float g = ag[mi][ni][r], u = au[mi][ni][r];
          float a = (g / (1.f + __expf(-g))) * u;
          act[((size_t)e * T + slot) * I + col] = tobf(a);
        }
      }
}

// ---------------------------------------------------------------------------
// MoE GEMM2: moe_out[token] += w_slot * (act[e] @ w2s[e]^T), atomic scatter.
__global__ __launch_bounds__(256) void moe_gemm2_kernel(
    const __bf16* __restrict__ act, const float* __restrict__ w2s,
    const int* __restrict__ counts, const int* __restrict__ list,
    const float* __restrict__ slotw, float* __restrict__ out) {
  int e = blockIdx.z;
  int cnt = counts[e];
  int m0b = blockIdx.x * 64;
  if (m0b >= cnt) return;
  int n0b = blockIdx.y * 128;
  __shared__ __bf16 As[2][64 * 32];
  __shared__ __bf16 Bs[2][128 * 32];
  __shared__ int rowTok[64];
  __shared__ float rowW[64];
  int tid = threadIdx.x, lane = tid & 31, wid = tid >> 5;
  if (tid < 64) {
    int s = m0b + tid;
    int cs = (s < cnt) ? s : 0;
    rowTok[tid] = list[e * T + cs];
    rowW[tid] = slotw[e * T + cs];
  }
  __syncthreads();
  int wm = (wid & 1) * 32, wn = (wid >> 1) * 32;
  const __bf16* Arow = act + (size_t)e * T * I + (size_t)m0b * I;
  const float*  Brow = w2s + (size_t)e * H * I + (size_t)n0b * I;
  v8f acc[2][2] = {};
  float4 breg[4];
  load_b_tile(Brow, I, 0, tid, breg);
  int p = 0;
  for (int kk = 0; kk < I; kk += 32, p ^= 1) {
    copy_a_tile(Arow, I, kk, tid, As[p]);
    store_b_tile(Bs[p], tid, breg);
    if (kk + 32 < I) load_b_tile(Brow, I, kk + 32, tid, breg);
    block_sync_lds();
    v16bf a0 = lds_afrag(&As[p][wm * 32], 32, lane);
    v16bf a1 = lds_afrag(&As[p][(wm + 16) * 32], 32, lane);
    v16bf b0 = lds_bfrag(&Bs[p][wn * 32], 32, lane);
    v16bf b1 = lds_bfrag(&Bs[p][(wn + 16) * 32], 32, lane);
    acc[0][0] = wmma_bf16(a0, b0, acc[0][0]);
    acc[0][1] = wmma_bf16(a0, b1, acc[0][1]);
    acc[1][0] = wmma_bf16(a1, b0, acc[1][0]);
    acc[1][1] = wmma_bf16(a1, b1, acc[1][1]);
  }
  int half = lane >> 4, nidx = lane & 15;
#pragma unroll
  for (int mi = 0; mi < 2; mi++)
#pragma unroll
    for (int ni = 0; ni < 2; ni++)
#pragma unroll
      for (int r = 0; r < 8; r++) {
        int sl = wm + mi * 16 + r + half * 8;        // local slot (0..63)
        if (m0b + sl < cnt) {
          int col = n0b + wn + ni * 16 + nidx;
          atomicAdd(&out[(size_t)rowTok[sl] * H + col], rowW[sl] * acc[mi][ni][r]);
        }
      }
}

// ---------------------------------------------------------------------------
extern "C" void kernel_launch(void* const* d_in, const int* in_sizes, int n_in,
                              void* d_out, int out_size, void* d_ws,
                              size_t ws_size, hipStream_t stream) {
  const float* hs   = (const float*)d_in[0];
  // d_in[1] = positions (unused by the reference math)
  const float* ln1w = (const float*)d_in[2];
  const float* qkvw = (const float*)d_in[3];
  const float* ow   = (const float*)d_in[4];
  const float* ln2w = (const float*)d_in[5];
  const float* rw   = (const float*)d_in[6];
  const float* wsw  = (const float*)d_in[7];
  const float* w2s  = (const float*)d_in[8];
  float* out_moe = (float*)d_out;
  float* out_res = out_moe + (size_t)T * H;

  char* ws = (char*)d_ws;
  auto aln = [](size_t x) { return (x + 255) & ~(size_t)255; };
  size_t off = 0;
  __bf16* h1    = (__bf16*)(ws + off); off = aln(off + (size_t)T * H * 2);
  float*  qkv   = (float*)(ws + off);  off = aln(off + (size_t)T * QKVD * 4);
  __bf16* attnb = (__bf16*)(ws + off); off = aln(off + (size_t)T * H * 2);
  float*  attno = (float*)(ws + off);  off = aln(off + (size_t)T * H * 4);
  __bf16* h2    = (__bf16*)(ws + off); off = aln(off + (size_t)T * H * 2);
  int*    counts= (int*)(ws + off);    off = aln(off + (size_t)E * 4);
  int*    list  = (int*)(ws + off);    off = aln(off + (size_t)E * T * 4);
  float*  slotw = (float*)(ws + off);  off = aln(off + (size_t)E * T * 4);
  __bf16* act   = (__bf16*)(ws + off); off = aln(off + (size_t)E * T * I * 2);

  init_kernel<<<dim3((T * H + 255) / 256), 256, 0, stream>>>(out_moe, counts);
  rmsnorm1_kernel<<<dim3(T), 256, 0, stream>>>(hs, ln1w, h1);
  gemm_kernel<<<dim3(T / 64, QKVD / 128), 256, 0, stream>>>(h1, qkvw, qkv, T, QKVD, H);
  attn_kernel<<<dim3(T / 64, NH), 128, 0, stream>>>(qkv, attnb);
  gemm_kernel<<<dim3(T / 64, H / 128), 256, 0, stream>>>(attnb, ow, attno, T, H, H);
  add_rms_kernel<<<dim3(T), 256, 0, stream>>>(hs, attno, ln2w, out_res, h2);
  router_kernel<<<dim3(T / 8), 256, 0, stream>>>(h2, rw, counts, list, slotw);
  moe_gemm1_kernel<<<dim3(T / 64, I / 128, E), 256, 0, stream>>>(h2, wsw, counts, list, act);
  moe_gemm2_kernel<<<dim3(T / 64, H / 128, E), 256, 0, stream>>>(act, w2s, counts, list, slotw, out_moe);
}
